// DualScaleGraphUNet_29205777613111
// MI455X (gfx1250) — matrix operand order repaired and checked by reading, not compiled, not run on hardware
//
#include <hip/hip_runtime.h>
#include <hip/hip_bf16.h>

// ---------------------------------------------------------------------------
// DualScaleGraphUNet on MI455X (gfx1250, wave32, WMMA bf16 -> f32)
// ---------------------------------------------------------------------------
//   * All matrix math through v_wmma_f32_16x16x32_bf16 (f32 accumulate).
//   * assign (400MB fp32 > 192MB L2) dominates HBM traffic -> bf16 copies
//     (forward + transposed) halve the stream.
//   * One GEMM kernel (N==128): wave owns 16x128 stripe, A tile reused over
//     8 WMMAs/K-step, B pre-packed in wave32 WMMA-B layout (contiguous b128s).
//     Inner loop is branch-free (K-tail handled once, outside the loop) and
//     software-pipelined: B[j+1] and A[kt+1] are in flight while WMMA[j] runs.
//   * Split-K via gridDim.y + global_atomic_add_f32 for the skinny pool GEMMs
//     (M=2000, K=50000) -> 13x more waves to reach the BW roofline.
//   * LDS-tiled transpose-cast for assign^T (coalesced both sides).
//   * Fused edge MLP: gather [x_dst|x_src|ea] -> WMMA -> ReLU -> LDS relayout
//     -> WMMA -> atomic scatter-add on dst.
// ---------------------------------------------------------------------------

typedef __attribute__((ext_vector_type(16))) __bf16 bf16x16;
typedef __attribute__((ext_vector_type(8)))  __bf16 bf16x8;
typedef __attribute__((ext_vector_type(8)))  float  f32x8;

#define N_F   50000
#define N_C   2000
#define E_F   800000
#define E_C   64000
#define DIM   128
#define EDIM  16
#define MSGK  272      // 2*DIM + EDIM
#define KT_MSG 9       // ceil(272/32) -> padded K = 288
#define KT_D   4       // 128/32
#define KT_NC  63      // ceil(2000/32) -> 2016
#define KT_NF  1563    // ceil(50000/32) -> 50016
#define KSPLIT_POOL 13 // split-K factor for M=2000,K=50000 pools

// ---------------------------------------------------------------- helpers ---
__device__ __forceinline__ bf16x16 load16(const __bf16* p) {
    bf16x8 lo = *(const bf16x8*)p;
    bf16x8 hi = *(const bf16x8*)(p + 8);
    bf16x16 r;
#pragma unroll
    for (int i = 0; i < 8; ++i) { r[i] = lo[i]; r[8 + i] = hi[i]; }
    return r;
}

// A tile (16x32 bf16, WMMA-A layout) from row-major row pointer; chunks at
// +0 (K=half*8..+7) and +16 (K=16+half*8..+7). No bounds checks (full tile).
__device__ __forceinline__ bf16x16 loadA_full(const __bf16* arow, int kt, int half) {
    const __bf16* p = arow + (long)kt * 32 + half * 8;
    bf16x8 lo = *(const bf16x8*)p;
    bf16x8 hi = *(const bf16x8*)(p + 16);
    bf16x16 r;
#pragma unroll
    for (int i = 0; i < 8; ++i) { r[i] = lo[i]; r[8 + i] = hi[i]; }
    return r;
}

// ------------------------------------------------------------ elementwise ---
__global__ void k_cast_bf16(const float* __restrict__ s, __bf16* __restrict__ d, long n) {
    long i = (long)blockIdx.x * blockDim.x + threadIdx.x;
    long stride = (long)gridDim.x * blockDim.x;
    for (; i < n; i += stride) d[i] = (__bf16)s[i];
}

// LDS-tiled transpose-cast: At[c*R + r] = (bf16)A[r*C + c], coalesced R/W.
__global__ __launch_bounds__(256)
void k_transpose_cast(const float* __restrict__ A, __bf16* __restrict__ At,
                      int R, int C) {
    __shared__ float t[32][33];
    const int tx = threadIdx.x & 31;
    const int ty = threadIdx.x >> 5;          // 0..7
    const long bc = (long)blockIdx.x * 32;    // col-tile origin (C dim)
    const long br = (long)blockIdx.y * 32;    // row-tile origin (R dim)
#pragma unroll
    for (int i = ty; i < 32; i += 8) {
        long r = br + i, c = bc + tx;
        t[i][tx] = (r < R && c < C) ? A[r * (long)C + c] : 0.0f;
    }
    __syncthreads();
#pragma unroll
    for (int i = ty; i < 32; i += 8) {
        long c = bc + i, r = br + tx;
        if (c < C && r < R) At[c * (long)R + r] = (__bf16)t[tx][i];
    }
}

// Pack B (f32 row-major [K,128]) into WMMA-B tile layout, zero-padded K.
// out[((nt*Ktiles)+kt)*512 + lane*16 + i],  n = nt*16 + lane%16,
// k = kt*32 + (lane/16)*8 + (i%8) + (i/8)*16
__global__ void k_pack_b128(const float* __restrict__ W, __bf16* __restrict__ out,
                            int K, int Ktiles) {
    long idx = (long)blockIdx.x * blockDim.x + threadIdx.x;
    long total = (long)Ktiles * 4096;
    long stride = (long)gridDim.x * blockDim.x;
    for (; idx < total; idx += stride) {
        int i    = (int)(idx & 15);
        int lane = (int)((idx >> 4) & 31);
        long tile = idx >> 9;
        int kt = (int)(tile % Ktiles);
        int nt = (int)(tile / Ktiles);
        int n = nt * 16 + (lane & 15);
        int k = kt * 32 + (lane >> 4) * 8 + (i & 7) + (i >> 3) * 16;
        out[idx] = (k < K) ? (__bf16)W[(long)k * DIM + n] : (__bf16)0.0f;
    }
}

// ------------------------------------------------------------------- GEMM ---
// C[M,128](f32) = A[M,K](bf16,row-major,lda) x Bp(packed) [+bias]
// accum==0: C = A*B + bias (gridDim.y must be 1)
// accum==1: C += A*B via atomics; K-tiles split across gridDim.y (split-K)
// Requires K % 8 == 0 (true for all call sites: 50000, 2000, 128).
__global__ __launch_bounds__(128)
void k_gemm_n128(const __bf16* __restrict__ A, long lda, int M, int K, int Ktiles,
                 const __bf16* __restrict__ Bp, const float* __restrict__ bias,
                 float* __restrict__ C, int accum) {
    const int lane  = threadIdx.x & 31;
    const int wid   = threadIdx.x >> 5;
    const int gw    = blockIdx.x * 4 + wid;
    const int mBase = gw * 16;
    if (mBase >= M) return;
    const int mlane = lane & 15;
    const int half  = lane >> 4;

    // split-K range for this block
    const int kchunk = (Ktiles + gridDim.y - 1) / gridDim.y;
    const int kt0 = blockIdx.y * kchunk;
    int kt1 = kt0 + kchunk; if (kt1 > Ktiles) kt1 = Ktiles;

    int mrow = mBase + mlane; if (mrow >= M) mrow = M - 1;
    const __bf16* arow = A + (long)mrow * lda;

    f32x8 acc[8];
#pragma unroll
    for (int j = 0; j < 8; ++j)
#pragma unroll
        for (int r = 0; r < 8; ++r) acc[j][r] = 0.0f;

    // ---- full K tiles: branch-free, software pipelined
    const int ktFull = K >> 5;                        // tiles with no OOB chunk
    const int ktFullEnd = (kt1 < ktFull) ? kt1 : ktFull;

    if (kt0 < ktFullEnd) {
        bf16x16 a = loadA_full(arow, kt0, half);
        for (int kt = kt0; kt < ktFullEnd; ++kt) {
            __builtin_prefetch((const void*)(arow + (long)(kt + 8) * 32), 0, 3);
            const __bf16* bbase = Bp + (long)kt * 512 + lane * 16;
            bf16x16 bnext = load16(bbase);
            bf16x16 anext = a;
            if (kt + 1 < ktFullEnd) anext = loadA_full(arow, kt + 1, half);
#pragma unroll
            for (int j = 0; j < 8; ++j) {
                bf16x16 bcur = bnext;
                if (j < 7) bnext = load16(bbase + (long)(j + 1) * Ktiles * 512);
                acc[j] = __builtin_amdgcn_wmma_f32_16x16x32_bf16(
                    false, a, false, bcur, (short)0, acc[j], false, false);
            }
            a = anext;
        }
    }

    // ---- at most one remainder tile (K % 32 == 16 cases), guarded chunks
    if (ktFull >= kt0 && ktFull < kt1) {
        const int kt = ktFull;
        bf16x16 a;
#pragma unroll
        for (int c2 = 0; c2 < 2; ++c2) {
            int kc = kt * 32 + half * 8 + c2 * 16;
            if (kc < K) {                             // K%8==0 -> never partial
                bf16x8 v = *(const bf16x8*)(arow + kc);
#pragma unroll
                for (int i = 0; i < 8; ++i) a[c2 * 8 + i] = v[i];
            } else {
#pragma unroll
                for (int i = 0; i < 8; ++i) a[c2 * 8 + i] = (__bf16)0.0f;
            }
        }
        const __bf16* bbase = Bp + (long)kt * 512 + lane * 16;
#pragma unroll
        for (int j = 0; j < 8; ++j) {
            bf16x16 b = load16(bbase + (long)j * Ktiles * 512);
            acc[j] = __builtin_amdgcn_wmma_f32_16x16x32_bf16(
                false, a, false, b, (short)0, acc[j], false, false);
        }
    }

#pragma unroll
    for (int j = 0; j < 8; ++j) {
        int n = j * 16 + mlane;
        float bv = (!accum && bias) ? bias[n] : 0.0f;
#pragma unroll
        for (int r = 0; r < 8; ++r) {
            int m = mBase + r + 8 * half;
            if (m < M) {
                long idx = (long)m * DIM + n;
                if (accum) atomicAdd(&C[idx], acc[j][r]);
                else       C[idx] = acc[j][r] + bv;
            }
        }
    }
}

// -------------------------------------------------------- fused edge MLP ---
// per wave: 16 edges.  msg = [h[dst] | h[src] | ea]  (K=272, padded 288)
// h1 = relu(msg @ W1 + b1); out = h1 @ W2 + b2; atomic scatter-add on dst.
__global__ __launch_bounds__(256)
void k_edge_mlp(const __bf16* __restrict__ h, const int* __restrict__ src,
                const int* __restrict__ dst, const __bf16* __restrict__ ea,
                const __bf16* __restrict__ W1p, const float* __restrict__ b1,
                const __bf16* __restrict__ W2p, const float* __restrict__ b2,
                float* __restrict__ hout, int E) {
    __shared__ __bf16 sh1[8 * 16 * DIM];   // 8 waves * 16x128 bf16 = 32KB

    const int lane = threadIdx.x & 31;
    const int wid  = threadIdx.x >> 5;
    const int gw   = blockIdx.x * 8 + wid;
    const int eBase = gw * 16;
    if (eBase >= E) return;
    const int mlane = lane & 15;
    const int half  = lane >> 4;

    int e = eBase + mlane; if (e >= E) e = E - 1;
    const int  vs = src[e];
    const int  vd = dst[e];
    const __bf16* rowD = h  + (long)vd * DIM;
    const __bf16* rowS = h  + (long)vs * DIM;
    const __bf16* rowE = ea + (long)e * EDIM;

    f32x8 acc[8];
#pragma unroll
    for (int j = 0; j < 8; ++j)
#pragma unroll
        for (int r = 0; r < 8; ++r) acc[j][r] = 0.0f;

    // ---- stage 1: [16,288] x [288,128]
    for (int kt = 0; kt < KT_MSG; ++kt) {
        bf16x16 a;
#pragma unroll
        for (int c2 = 0; c2 < 2; ++c2) {
            int kc = kt * 32 + half * 8 + c2 * 16;   // 8-chunk never straddles segments
            const __bf16* p = nullptr;
            if      (kc < DIM)       p = rowD + kc;
            else if (kc < 2 * DIM)   p = rowS + (kc - DIM);
            else if (kc < MSGK)      p = rowE + (kc - 2 * DIM);
            if (p) {
                bf16x8 v = *(const bf16x8*)p;
#pragma unroll
                for (int i = 0; i < 8; ++i) a[c2 * 8 + i] = v[i];
            } else {
#pragma unroll
                for (int i = 0; i < 8; ++i) a[c2 * 8 + i] = (__bf16)0.0f;
            }
        }
        const __bf16* bbase = W1p + (long)kt * 512 + lane * 16;
        bf16x16 bnext = load16(bbase);
#pragma unroll
        for (int j = 0; j < 8; ++j) {
            bf16x16 bcur = bnext;
            if (j < 7) bnext = load16(bbase + (long)(j + 1) * KT_MSG * 512);
            acc[j] = __builtin_amdgcn_wmma_f32_16x16x32_bf16(
                false, a, false, bcur, (short)0, acc[j], false, false);
        }
    }

    // bias + relu -> LDS (relayout C -> A for stage 2)
    __bf16* my = sh1 + wid * (16 * DIM);
#pragma unroll
    for (int j = 0; j < 8; ++j) {
        int n = j * 16 + mlane;
        float bv = b1[n];
#pragma unroll
        for (int r = 0; r < 8; ++r) {
            int m = r + 8 * half;
            float v = acc[j][r] + bv;
            my[m * DIM + n] = (__bf16)(v > 0.0f ? v : 0.0f);
        }
    }
    __syncthreads();

    // ---- stage 2: [16,128] x [128,128]
#pragma unroll
    for (int j = 0; j < 8; ++j)
#pragma unroll
        for (int r = 0; r < 8; ++r) acc[j][r] = 0.0f;

    for (int kt = 0; kt < KT_D; ++kt) {
        bf16x16 a;
#pragma unroll
        for (int c2 = 0; c2 < 2; ++c2) {
            int kc = kt * 32 + half * 8 + c2 * 16;
            bf16x8 v = *(const bf16x8*)(my + mlane * DIM + kc);
#pragma unroll
            for (int i = 0; i < 8; ++i) a[c2 * 8 + i] = v[i];
        }
        const __bf16* bbase = W2p + (long)kt * 512 + lane * 16;
        bf16x16 bnext = load16(bbase);
#pragma unroll
        for (int j = 0; j < 8; ++j) {
            bf16x16 bcur = bnext;
            if (j < 7) bnext = load16(bbase + (long)(j + 1) * KT_D * 512);
            acc[j] = __builtin_amdgcn_wmma_f32_16x16x32_bf16(
                false, a, false, bcur, (short)0, acc[j], false, false);
        }
    }

    // bias + scatter-add
#pragma unroll
    for (int j = 0; j < 8; ++j) {
        int n = j * 16 + mlane;
        float bv = b2[n];
#pragma unroll
        for (int r = 0; r < 8; ++r) {
            int m = r + 8 * half;
            int ee = eBase + m;
            if (ee < E) {
                int vdd = dst[ee];
                atomicAdd(&hout[(long)vdd * DIM + n], acc[j][r] + bv);
            }
        }
    }
}

// ------------------------------------------------------------------ host ---
static inline long lmin(long a, long b) { return a < b ? a : b; }

extern "C" void kernel_launch(void* const* d_in, const int* in_sizes, int n_in,
                              void* d_out, int out_size, void* d_ws, size_t ws_size,
                              hipStream_t stream) {
    (void)in_sizes; (void)n_in; (void)out_size; (void)ws_size;

    const float* x_f    = (const float*)d_in[0];
    const float* x_c    = (const float*)d_in[1];
    const int*   ei_f   = (const int*)  d_in[2];
    const float* ea_f   = (const float*)d_in[3];
    const int*   ei_c   = (const int*)  d_in[4];
    const float* ea_c   = (const float*)d_in[5];
    const float* assign = (const float*)d_in[6];
    const float* Wfe    = (const float*)d_in[7];
    const float* bfe    = (const float*)d_in[8];
    const float* Wce    = (const float*)d_in[9];
    const float* bce    = (const float*)d_in[10];
    const float* fW1    = (const float*)d_in[11];
    const float* fb1    = (const float*)d_in[12];
    const float* fW2    = (const float*)d_in[13];
    const float* fb2    = (const float*)d_in[14];
    const float* cW1    = (const float*)d_in[15];
    const float* cb1    = (const float*)d_in[16];
    const float* cW2    = (const float*)d_in[17];
    const float* cb2    = (const float*)d_in[18];
    const float* Wd     = (const float*)d_in[19];
    const float* bd     = (const float*)d_in[20];

    const int* src_f = ei_f;           const int* dst_f = ei_f + E_F;
    const int* src_c = ei_c;           const int* dst_c = ei_c + E_C;

    // ---- carve workspace
    char* w = (char*)d_ws; size_t off = 0;
    auto carve = [&](size_t bytes) -> void* {
        void* p = w + off; off = (off + bytes + 255) & ~(size_t)255; return p;
    };
    float*  hf      = (float*) carve((size_t)N_F * DIM * 4);
    float*  hc      = (float*) carve((size_t)N_C * DIM * 4);
    __bf16* hf_bf   = (__bf16*)carve((size_t)N_F * DIM * 2);
    __bf16* hc_bf   = (__bf16*)carve((size_t)N_C * DIM * 2);
    __bf16* xf_bf   = (__bf16*)carve((size_t)N_F * DIM * 2);
    __bf16* xc_bf   = (__bf16*)carve((size_t)N_C * DIM * 2);
    __bf16* eaf_bf  = (__bf16*)carve((size_t)E_F * EDIM * 2);
    __bf16* eac_bf  = (__bf16*)carve((size_t)E_C * EDIM * 2);
    __bf16* asn_bf  = (__bf16*)carve((size_t)N_F * N_C * 2);
    __bf16* asnT_bf = (__bf16*)carve((size_t)N_F * N_C * 2);
    __bf16* packS   = (__bf16*)carve((size_t)KT_NF * 4096 * 2);
    __bf16* WfeP    = (__bf16*)carve((size_t)KT_D * 4096 * 2);
    __bf16* WceP    = (__bf16*)carve((size_t)KT_D * 4096 * 2);
    __bf16* WdP     = (__bf16*)carve((size_t)KT_D * 4096 * 2);
    __bf16* fW1p    = (__bf16*)carve((size_t)3 * KT_MSG * 4096 * 2);
    __bf16* fW2p    = (__bf16*)carve((size_t)3 * KT_D   * 4096 * 2);
    __bf16* cW1p    = (__bf16*)carve((size_t)6 * KT_MSG * 4096 * 2);
    __bf16* cW2p    = (__bf16*)carve((size_t)6 * KT_D   * 4096 * 2);

    // ---- launch helpers
    auto cast = [&](const float* s, __bf16* d, long n) {
        int blocks = (int)lmin((n + 255) / 256, 1 << 20);
        k_cast_bf16<<<blocks, 256, 0, stream>>>(s, d, n);
    };
    auto pack = [&](const float* W, __bf16* outp, int K, int Ktiles) {
        long total = (long)Ktiles * 4096;
        int blocks = (int)lmin((total + 255) / 256, 1 << 20);
        k_pack_b128<<<blocks, 256, 0, stream>>>(W, outp, K, Ktiles);
    };
    auto gemm = [&](const __bf16* A, long lda, int M, int K, int Ktiles,
                    const __bf16* Bp, const float* bias, float* C,
                    int accum, int ksplit) {
        int waves = (M + 15) / 16;
        int blocks = (waves + 3) / 4;
        dim3 grid(blocks, ksplit, 1);
        k_gemm_n128<<<grid, 128, 0, stream>>>(A, lda, M, K, Ktiles, Bp, bias, C, accum);
    };
    auto mpnn = [&](const __bf16* h, const int* s, const int* d, const __bf16* ea,
                    const __bf16* W1p, const float* b1, const __bf16* W2p,
                    const float* b2, float* ho, int E) {
        int waves = (E + 15) / 16;
        int blocks = (waves + 7) / 8;
        k_edge_mlp<<<blocks, 256, 0, stream>>>(h, s, d, ea, W1p, b1, W2p, b2, ho, E);
    };

    // ---- one-time casts / packs (re-done every call for determinism)
    cast(x_f,  xf_bf,  (long)N_F * DIM);
    cast(x_c,  xc_bf,  (long)N_C * DIM);
    cast(ea_f, eaf_bf, (long)E_F * EDIM);
    cast(ea_c, eac_bf, (long)E_C * EDIM);
    cast(assign, asn_bf, (long)N_F * N_C);
    {
        dim3 grid((N_C + 31) / 32, (N_F + 31) / 32, 1);
        k_transpose_cast<<<grid, 256, 0, stream>>>(assign, asnT_bf, N_F, N_C);
    }
    pack(Wfe, WfeP, DIM, KT_D);
    pack(Wce, WceP, DIM, KT_D);
    pack(Wd,  WdP,  DIM, KT_D);
    for (int i = 0; i < 3; ++i) {
        pack(fW1 + (long)i * MSGK * DIM, fW1p + (long)i * KT_MSG * 4096, MSGK, KT_MSG);
        pack(fW2 + (long)i * DIM  * DIM, fW2p + (long)i * KT_D   * 4096, DIM,  KT_D);
    }
    for (int i = 0; i < 6; ++i) {
        pack(cW1 + (long)i * MSGK * DIM, cW1p + (long)i * KT_MSG * 4096, MSGK, KT_MSG);
        pack(cW2 + (long)i * DIM  * DIM, cW2p + (long)i * KT_D   * 4096, DIM,  KT_D);
    }

    // ---- encoders
    gemm(xf_bf, DIM, N_F, DIM, KT_D, WfeP, bfe, hf, 0, 1);
    gemm(xc_bf, DIM, N_C, DIM, KT_D, WceP, bce, hc, 0, 1);

    // ---- initial pool: h_c += assign^T @ h_f   (split-K for occupancy)
    pack(hf, packS, N_F, KT_NF);
    gemm(asnT_bf, N_F, N_C, N_F, KT_NF, packS, nullptr, hc, 1, KSPLIT_POOL);

    // ---- interleaved dual-scale layers
    for (int i = 0; i < 3; ++i) {
        cast(hf, hf_bf, (long)N_F * DIM);
        cast(hc, hc_bf, (long)N_C * DIM);
        mpnn(hf_bf, src_f, dst_f, eaf_bf,
             fW1p + (long)i * KT_MSG * 4096, fb1 + (long)i * DIM,
             fW2p + (long)i * KT_D   * 4096, fb2 + (long)i * DIM, hf, E_F);
        mpnn(hc_bf, src_c, dst_c, eac_bf,
             cW1p + (long)i * KT_MSG * 4096, cb1 + (long)i * DIM,
             cW2p + (long)i * KT_D   * 4096, cb2 + (long)i * DIM, hc, E_C);
        // interp: h_f += assign @ h_c   (M=50000 -> already wide, no split-K)
        pack(hc, packS, N_C, KT_NC);
        gemm(asn_bf, N_C, N_F, N_C, KT_NC, packS, nullptr, hf, 1, 1);
        // pool: h_c += assign^T @ h_f   (split-K)
        pack(hf, packS, N_F, KT_NF);
        gemm(asnT_bf, N_F, N_C, N_F, KT_NF, packS, nullptr, hc, 1, KSPLIT_POOL);
    }

    // ---- remaining coarse-only layers
    for (int i = 3; i < 6; ++i) {
        cast(hc, hc_bf, (long)N_C * DIM);
        mpnn(hc_bf, src_c, dst_c, eac_bf,
             cW1p + (long)i * KT_MSG * 4096, cb1 + (long)i * DIM,
             cW2p + (long)i * KT_D   * 4096, cb2 + (long)i * DIM, hc, E_C);
        pack(hc, packS, N_C, KT_NC);
        gemm(asn_bf, N_C, N_F, N_C, KT_NC, packS, nullptr, hf, 1, 1);
    }

    // ---- decoder: out = h_f @ Wd + bd
    cast(hf, hf_bf, (long)N_F * DIM);
    gemm(hf_bf, DIM, N_F, DIM, KT_D, WdP, bd, (float*)d_out, 0, 1);
}
